// GOATSelfAttention_73083163508886
// MI455X (gfx1250) — matrix operand order repaired and checked by reading, not compiled, not run on hardware
//
#include <hip/hip_runtime.h>
#include <hip/hip_bf16.h>
#include <stdint.h>
#include <stddef.h>

// ---------------------------------------------------------------------------
// GOAT self-attention with Fourier relative bias, CDNA5 (gfx1250) WMMA path.
//   B=4, T=2048, D=1024, H=16, dh=64, R=32, causal.
// Matrix math on v_wmma_f32_16x16x32_bf16 (wave32); LDS staging through
// GLOBAL_LOAD_ASYNC_TO_LDS_B128 when the toolchain exposes the builtin.
// ---------------------------------------------------------------------------

typedef __attribute__((ext_vector_type(16))) __bf16 v16bf;
typedef __attribute__((ext_vector_type(8)))  __bf16 v8bf;
typedef __attribute__((ext_vector_type(8)))  float  v8f;

#define TSEQ   2048
#define DMODEL 1024
#define NHEADS 16
#define HDIM   64
#define NFREQ  32
#define QKVLD  3072
#define BIASW  (2*TSEQ - 1)

// ---------------------------------------------------------------------------
// Async global->LDS staging (CDNA5). Falls back to VGPR round-trip copies.
// Builtin signature (from probe diagnostics): param0 = int4 AS(1)* (global
// source, non-const), param1 = int4 AS(3)* (LDS dest), imm offset, imm cpol.
// ---------------------------------------------------------------------------
#if defined(__has_builtin)
#if __has_builtin(__builtin_amdgcn_global_load_async_to_lds_b128)
#define HAVE_ASYNC_LDS 1
#endif
#endif

#if defined(HAVE_ASYNC_LDS)
typedef int vsi4 __attribute__((vector_size(4 * sizeof(int))));
typedef __attribute__((address_space(1))) vsi4* gas_v4p;
typedef __attribute__((address_space(3))) vsi4* las_v4p;
__device__ __forceinline__ void stage16(const __bf16* g, __bf16* l) {
  __builtin_amdgcn_global_load_async_to_lds_b128((gas_v4p)(__bf16*)g,
                                                 (las_v4p)l, 0, 0);
}
__device__ __forceinline__ void stage_fence() {
#if __has_builtin(__builtin_amdgcn_s_wait_asynccnt)
  __builtin_amdgcn_s_wait_asynccnt(0);
#else
  asm volatile("s_wait_asynccnt 0x0" ::: "memory");
#endif
  __syncthreads();
}
#else
__device__ __forceinline__ void stage16(const __bf16* g, __bf16* l) {
  *(uint4*)l = *(const uint4*)g;
}
__device__ __forceinline__ void stage_fence() { __syncthreads(); }
#endif

// 64-byte staging helper (4 x b128 per thread).
__device__ __forceinline__ void stage64B(const __bf16* g, __bf16* l) {
  stage16(g, l);
  stage16(g + 8, l + 8);
  stage16(g + 16, l + 16);
  stage16(g + 24, l + 24);
}

// ---------------------------------------------------------------------------
// WMMA fragment loaders (bf16, 16x16x32, wave32) from row-major LDS tiles.
// A (16x32, MxK): lane L -> row m=L&15, half=L>>4:
//   elems 0..15 K = 16*half + 0..15 (32 contiguous bytes)
// B (32x16, KxN) stored [N][K]: lane L -> col n=L&15, half=L>>4:
//   elems 0..15 K = 16*half + 0..15 (32 contiguous bytes)
// ---------------------------------------------------------------------------
__device__ __forceinline__ v16bf frag_a(const __bf16* base, int ld, int m,
                                        int k0, int half) {
  return *(const v16bf*)(base + m * ld + k0 + 16 * half);
}

__device__ __forceinline__ v16bf frag_b(const __bf16* base, int ld, int n,
                                        int k0, int half) {
  return *(const v16bf*)(base + n * ld + k0 + 16 * half);
}

__device__ __forceinline__ v8f wmma_bf16(v16bf a, v16bf b, v8f c) {
  return __builtin_amdgcn_wmma_f32_16x16x32_bf16(false, a, false, b,
                                                 (short)0, c, false, false);
}

// ---------------------------------------------------------------------------
// Elementwise f32 -> bf16 conversion.
// ---------------------------------------------------------------------------
__global__ void cvt_f32_to_bf16(const float* __restrict__ in,
                                __bf16* __restrict__ out, int n) {
  int i = blockIdx.x * blockDim.x + threadIdx.x;
  if (i < n) out[i] = (__bf16)in[i];
}

// ---------------------------------------------------------------------------
// Fourier bias table: bias[h][d], displacement d-(T-1) in [-(T-1), T-1].
// ---------------------------------------------------------------------------
__global__ void fourier_bias_kernel(const float* __restrict__ alpha,
                                    const float* __restrict__ beta,
                                    float* __restrict__ bias) {
  int idx = blockIdx.x * blockDim.x + threadIdx.x;
  if (idx >= NHEADS * BIASW) return;
  int h = idx / BIASW;
  int d = idx % BIASW;
  float disp = (float)(d - (TSEQ - 1));
  const float ln_base = 9.210340371976184f;  // ln(10000)
  float acc = 0.0f;
#pragma unroll 4
  for (int r = 0; r < NFREQ; ++r) {
    float omega = __expf(-2.0f * (float)r / (float)HDIM * ln_base);
    float s, c;
    __sincosf(omega * disp, &s, &c);
    acc += alpha[h * NFREQ + r] * c + beta[h * NFREQ + r] * s;
  }
  bias[idx] = acc;
}

// ---------------------------------------------------------------------------
// Double-buffered WMMA GEMM: C[M][N] = A[M][K] * B[N][K]^T  (x @ W.T)
// 128x128 C tile, 256 threads = 8 waves (4M x 2N); wave owns 32x64 = 2x4
// accumulators. 64-wide K slices staged async into LDS ping-pong buffers;
// the fill of slice i+1 overlaps the 16 WMMAs of slice i.
// OUTMODE: 0 = f32 out, 1 = bf16 out, 2 = qkv split (q,k -> Cout row-major,
//          v -> vT pre-transposed per (b,h): vT[(b*16+h)*64+d][t]).
// ---------------------------------------------------------------------------
template <int OUTMODE>
__global__ void __launch_bounds__(256)
gemm_xwT(const __bf16* __restrict__ A, const __bf16* __restrict__ B,
         void* __restrict__ Cout, __bf16* __restrict__ vT,
         int M, int N, int K) {
  __shared__ __align__(32) __bf16 sA[2][128 * 64];
  __shared__ __align__(32) __bf16 sB[2][128 * 64];

  const int tid  = threadIdx.x;
  const int lane = tid & 31;
  const int wid  = tid >> 5;
  const int half = lane >> 4;
  const int l15  = lane & 15;

  const int bm = blockIdx.y * 128;
  const int bn = blockIdx.x * 128;
  const int wm = 32 * (wid & 3);
  const int wn = 64 * (wid >> 2);

  const int row = tid >> 1;          // 0..127
  const int c0  = (tid & 1) * 32;    // staging column (bf16 elements)

  v8f acc[2][4];
#pragma unroll
  for (int i = 0; i < 2; ++i)
#pragma unroll
    for (int j = 0; j < 4; ++j)
#pragma unroll
      for (int r = 0; r < 8; ++r) acc[i][j][r] = 0.0f;

  // Prologue: stage slice 0 into buffer 0.
  stage64B(A + (size_t)(bm + row) * K + c0, &sA[0][row * 64 + c0]);
  stage64B(B + (size_t)(bn + row) * K + c0, &sB[0][row * 64 + c0]);
  stage_fence();

  int buf = 0;
  for (int k0 = 0; k0 < K; k0 += 64) {
    if (k0 + 64 < K) {  // kick async fill of the next slice
      stage64B(A + (size_t)(bm + row) * K + k0 + 64 + c0,
               &sA[buf ^ 1][row * 64 + c0]);
      stage64B(B + (size_t)(bn + row) * K + k0 + 64 + c0,
               &sB[buf ^ 1][row * 64 + c0]);
    }
#pragma unroll
    for (int ks = 0; ks < 2; ++ks) {
      v16bf af[2], bfrag[4];
#pragma unroll
      for (int mt = 0; mt < 2; ++mt)
        af[mt] = frag_a(sA[buf], 64, wm + 16 * mt + l15, 32 * ks, half);
#pragma unroll
      for (int nt = 0; nt < 4; ++nt)
        bfrag[nt] = frag_b(sB[buf], 64, wn + 16 * nt + l15, 32 * ks, half);
#pragma unroll
      for (int mt = 0; mt < 2; ++mt)
#pragma unroll
        for (int nt = 0; nt < 4; ++nt)
          acc[mt][nt] = wmma_bf16(af[mt], bfrag[nt], acc[mt][nt]);
    }
    stage_fence();  // next slice landed AND everyone is done with `buf`
    buf ^= 1;
  }

  // C layout: element r of lane L -> row 16*mt + r + 8*(L>>4), col L&15.
#pragma unroll
  for (int mt = 0; mt < 2; ++mt)
#pragma unroll
    for (int nt = 0; nt < 4; ++nt)
#pragma unroll
      for (int r = 0; r < 8; ++r) {
        int rr = bm + wm + 16 * mt + r + 8 * half;
        int cc = bn + wn + 16 * nt + l15;
        float v = acc[mt][nt][r];
        if constexpr (OUTMODE == 0) {
          ((float*)Cout)[(size_t)rr * N + cc] = v;
        } else if constexpr (OUTMODE == 1) {
          ((__bf16*)Cout)[(size_t)rr * N + cc] = (__bf16)v;
        } else {
          if (cc < 2 * DMODEL) {
            ((__bf16*)Cout)[(size_t)rr * N + cc] = (__bf16)v;
          } else {
            int bb = rr >> 11, t = rr & (TSEQ - 1);      // TSEQ == 2048
            int hc = cc - 2 * DMODEL;                    // h*64 + d, 0..1023
            vT[((size_t)(bb << 10) + hc) * TSEQ + t] = (__bf16)v;
          }
        }
      }
}

// ---------------------------------------------------------------------------
// Fused flash attention with Fourier bias + causal mask.
// Block = 128 threads (4 waves); one (b, h, 64-query tile) per block.
// Wave w owns query rows 16w..16w+15. Streaming softmax; S stays in LDS.
// V arrives pre-transposed from the QKV GEMM, so all staging is b128-wide.
// ---------------------------------------------------------------------------
__global__ void __launch_bounds__(128)
attn_kernel(const __bf16* __restrict__ qkv, const __bf16* __restrict__ vT,
            const float* __restrict__ bias, __bf16* __restrict__ out) {
  __shared__ __align__(32) __bf16 Qs[64 * 64];   // [qrow][dh]
  __shared__ __align__(32) __bf16 Ks[64 * 64];   // [key][dh]   (B for Q.K^T)
  __shared__ __align__(32) __bf16 Vt[64 * 64];   // [dh][key]   (B for P.V)
  __shared__ __align__(32) float  Sb[64 * 64];   // logits tile
  __shared__ __align__(32) __bf16 Pb[64 * 64];   // softmax numerators (bf16)
  __shared__ float sf[64];                       // per-row rescale factor
  __shared__ float linv[64];                     // per-row 1/sum

  const int tid  = threadIdx.x;
  const int lane = tid & 31;
  const int wid  = tid >> 5;
  const int half = lane >> 4;
  const int l15  = lane & 15;

  const int qb    = blockIdx.x;
  const int qbase = qb * 64;
  const int b     = blockIdx.y >> 4;  // NHEADS == 16
  const int h     = blockIdx.y & 15;

  const float scale = 0.125f;         // 1/sqrt(64)
  const float* biasH = bias + (size_t)h * BIASW;

  const int srow = tid >> 1;          // staging row 0..63
  const int sc0  = (tid & 1) * 32;    // staging column 0 or 32

  // Stage Q tile (64x64 bf16); completion covered by first in-loop fence.
  stage64B(qkv + (size_t)(b * TSEQ + qbase + srow) * QKVLD + h * HDIM + sc0,
           &Qs[srow * 64 + sc0]);

  v8f oacc[4];
#pragma unroll
  for (int nt = 0; nt < 4; ++nt)
#pragma unroll
    for (int r = 0; r < 8; ++r) oacc[nt][r] = 0.0f;

  float m_i = -1e30f;   // running row max (threads 0..63 own row tid)
  float l_i = 0.0f;     // running row sum

  for (int kb = 0; kb <= qb; ++kb) {
    const int kbase = kb * 64;
    __syncthreads();  // previous iteration done reading Ks/Vt/Pb

    // Stage K tile [key][dh] and V^T tile [dh][key], 64B per thread each.
    stage64B(qkv + (size_t)(b * TSEQ + kbase + srow) * QKVLD + DMODEL +
                 h * HDIM + sc0,
             &Ks[srow * 64 + sc0]);
    stage64B(vT + ((size_t)(b << 10) + h * HDIM + srow) * TSEQ + kbase + sc0,
             &Vt[srow * 64 + sc0]);
    stage_fence();

    // S = Q K^T for this wave's 16-row band (contraction over dh = 64).
    v8f s[4];
#pragma unroll
    for (int nt = 0; nt < 4; ++nt)
#pragma unroll
      for (int r = 0; r < 8; ++r) s[nt][r] = 0.0f;
#pragma unroll
    for (int ks = 0; ks < 2; ++ks) {
      v16bf a = frag_a(Qs, 64, 16 * wid + l15, 32 * ks, half);
#pragma unroll
      for (int nt = 0; nt < 4; ++nt) {
        v16bf bb = frag_b(Ks, 64, 16 * nt + l15, 32 * ks, half);
        s[nt] = wmma_bf16(a, bb, s[nt]);
      }
    }
#pragma unroll
    for (int nt = 0; nt < 4; ++nt)
#pragma unroll
      for (int r = 0; r < 8; ++r)
        Sb[(16 * wid + r + 8 * half) * 64 + 16 * nt + l15] = s[nt][r];
    __syncthreads();

    // Online softmax: threads 0..63 each own one query row.
    if (tid < 64) {
      const int m = tid;
      const int dbase = (qbase + m) - kbase + (TSEQ - 1);
      const bool full = (kb < qb);  // strictly-below-diagonal: all valid
      float rmax = -1e30f;
      for (int n = 0; n < 64; ++n) {
        float v = Sb[m * 64 + n] * scale + biasH[dbase - n];
        if (!full && n > m) v = -1e30f;
        rmax = fmaxf(rmax, v);
      }
      float newm = fmaxf(m_i, rmax);
      float f = __expf(m_i - newm);
      float lsum = l_i * f;
      for (int n = 0; n < 64; ++n) {
        float p = 0.0f;
        if (full || n <= m) {
          float v = Sb[m * 64 + n] * scale + biasH[dbase - n];
          p = __expf(v - newm);
        }
        lsum += p;
        Pb[m * 64 + n] = (__bf16)p;
      }
      m_i = newm;
      l_i = lsum;
      sf[m] = f;
    }
    __syncthreads();

    // Rescale running output, then O += P.V (contraction over 64 keys).
#pragma unroll
    for (int nt = 0; nt < 4; ++nt)
#pragma unroll
      for (int r = 0; r < 8; ++r)
        oacc[nt][r] *= sf[16 * wid + r + 8 * half];
#pragma unroll
    for (int ks = 0; ks < 2; ++ks) {
      v16bf a = frag_a(Pb, 64, 16 * wid + l15, 32 * ks, half);
#pragma unroll
      for (int nt = 0; nt < 4; ++nt) {
        v16bf bb = frag_b(Vt, 64, 16 * nt + l15, 32 * ks, half);
        oacc[nt] = wmma_bf16(a, bb, oacc[nt]);
      }
    }
  }

  if (tid < 64) linv[tid] = 1.0f / l_i;
  __syncthreads();

  // Normalized output, bf16, laid out [b*T + t][h*64 + dh].
#pragma unroll
  for (int nt = 0; nt < 4; ++nt)
#pragma unroll
    for (int r = 0; r < 8; ++r) {
      int rrow = 16 * wid + r + 8 * half;
      int c = 16 * nt + l15;
      float v = oacc[nt][r] * linv[rrow];
      out[(size_t)(b * TSEQ + qbase + rrow) * DMODEL + h * HDIM + c] = (__bf16)v;
    }
}

// ---------------------------------------------------------------------------
// Host-side orchestration.
// ---------------------------------------------------------------------------
extern "C" void kernel_launch(void* const* d_in, const int* in_sizes, int n_in,
                              void* d_out, int out_size, void* d_ws,
                              size_t ws_size, hipStream_t stream) {
  const float* x     = (const float*)d_in[0];
  const float* Wqkv  = (const float*)d_in[1];
  const float* Wout  = (const float*)d_in[2];
  const float* alpha = (const float*)d_in[3];
  const float* beta  = (const float*)d_in[4];

  const int B = in_sizes[0] / (TSEQ * DMODEL);   // 4
  const int M = B * TSEQ;                        // 8192 token rows

  // Workspace layout (bytes). attnb reuses the xb region (dead after GEMM1).
  char* ws = (char*)d_ws;
  __bf16* xb    = (__bf16*)(ws);                        // 16 MB, M*1024
  __bf16* attnb = (__bf16*)(ws);                        // reuse of xb
  __bf16* wqkvb = (__bf16*)(ws + (size_t)16777216);     //  6 MB, 3072*1024
  __bf16* woutb = (__bf16*)(ws + (size_t)23068672);     //  2 MB, 1024*1024
  __bf16* qkvb  = (__bf16*)(ws + (size_t)25165824);     // 48 MB, M*3072 (v third unused)
  __bf16* vTb   = (__bf16*)(ws + (size_t)75497472);     // 16 MB, B*16*64 x 2048
  float*  biasT = (float*)(ws + (size_t)92274688);      // 262 KB, 16*4095

  // 1) f32 -> bf16 conversions.
  {
    int n = M * DMODEL;
    cvt_f32_to_bf16<<<(n + 255) / 256, 256, 0, stream>>>(x, xb, n);
    n = 3 * DMODEL * DMODEL;
    cvt_f32_to_bf16<<<(n + 255) / 256, 256, 0, stream>>>(Wqkv, wqkvb, n);
    n = DMODEL * DMODEL;
    cvt_f32_to_bf16<<<(n + 255) / 256, 256, 0, stream>>>(Wout, woutb, n);
  }

  // 2) Fourier bias table: H x (2T-1).
  {
    int n = NHEADS * BIASW;
    fourier_bias_kernel<<<(n + 255) / 256, 256, 0, stream>>>(alpha, beta, biasT);
  }

  // 3) QKV projection (8192x3072); V lands pre-transposed in vTb.
  {
    dim3 grid(3 * DMODEL / 128, M / 128);
    gemm_xwT<2><<<grid, 256, 0, stream>>>(xb, wqkvb, qkvb, vTb,
                                          M, 3 * DMODEL, DMODEL);
  }

  // 4) Fused causal attention with Fourier bias.
  {
    dim3 grid(TSEQ / 64, B * NHEADS);
    attn_kernel<<<grid, 128, 0, stream>>>(qkvb, vTb, biasT, attnb);
  }

  // 5) Output projection straight into d_out (f32).
  {
    dim3 grid(DMODEL / 128, M / 128);
    gemm_xwT<0><<<grid, 256, 0, stream>>>(attnb, woutb, d_out, nullptr,
                                          M, DMODEL, DMODEL);
  }
}
